// LCL_42425686950476
// MI455X (gfx1250) — matrix operand order repaired and checked by reading, not compiled, not run on hardware
//
#include <hip/hip_runtime.h>
#include <hip/hip_bf16.h>
#include <math.h>

// Problem constants (fixed by the reference harness).
constexpr int   B_ = 4096;   // batch
constexpr int   D_ = 1024;   // feature dim
constexpr int   C_ = 32;     // classes
constexpr float TEMP = 0.07f;

typedef __attribute__((ext_vector_type(16))) _Float16 v16h;
typedef __attribute__((ext_vector_type(8)))  _Float16 v8h;
typedef __attribute__((ext_vector_type(8)))  float    v8f;

// ---------------------------------------------------------------------------
// Prep 1: zero the label histogram.
__global__ void k_zero(int* __restrict__ cnt) {
    int t = threadIdx.x;
    if (t < C_) cnt[t] = 0;
}

// Prep 2: split f32 features into f16 hi + f16 residual lo (one-time cost,
// removes all cvt VALU from the WMMA hot loop and halves its load bytes).
__global__ void k_convert(const float* __restrict__ F,
                          _Float16* __restrict__ Fhi, _Float16* __restrict__ Flo) {
    int t = blockIdx.x * blockDim.x + threadIdx.x;
    if (t < B_ * D_) {
        float x = F[t];
        _Float16 h = (_Float16)x;
        Fhi[t] = h;
        Flo[t] = (_Float16)(x - (float)h);
    }
}

// Prep 3: row softmax of weights (B x 32, tiny).
__global__ void k_softmax(const float* __restrict__ Wt, float* __restrict__ Ww) {
    int i = blockIdx.x * blockDim.x + threadIdx.x;
    if (i >= B_) return;
    const float* r = Wt + (size_t)i * C_;
    float m = -3.402823e38f;
#pragma unroll
    for (int c = 0; c < C_; ++c) m = fmaxf(m, r[c]);
    float e[C_];
    float s = 0.f;
#pragma unroll
    for (int c = 0; c < C_; ++c) { e[c] = __expf(r[c] - m); s += e[c]; }
    float inv = 1.0f / s;
    float* o = Ww + (size_t)i * C_;
#pragma unroll
    for (int c = 0; c < C_; ++c) o[c] = e[c] * inv;
}

// Prep 4: diagonal stabilizer m̂_i = ||f_i||^2 / T (one wave per row) and
// label histogram. The max-shift cancels analytically in log_prob, so the
// diagonal (the max up to vanishing probability) is a valid stabilizer.
__global__ void k_diag(const float* __restrict__ F, const int* __restrict__ labels,
                       float* __restrict__ mhat, int* __restrict__ cnt) {
    int wid  = (blockIdx.x * blockDim.x + threadIdx.x) >> 5;
    int lane = threadIdx.x & 31;
    if (wid >= B_) return;
    const float* r = F + (size_t)wid * D_;
    float s = 0.f;
    for (int k = lane; k < D_; k += 32) { float x = r[k]; s += x * x; }
#pragma unroll
    for (int o = 16; o >= 1; o >>= 1) s += __shfl_xor(s, o, 32);
    if (lane == 0) {
        mhat[wid] = s * (1.0f / TEMP);
        atomicAdd(&cnt[labels[wid]], 1);
    }
}

// Pack two 8-half runs into a 16-half WMMA fragment register set
// (folds into the load registers; no moves in practice).
__device__ inline v16h packAB(v8h a, v8h b) {
    v16h o;
#pragma unroll
    for (int e = 0; e < 8; ++e) { o[e] = a[e]; o[8 + e] = b[e]; }
    return o;
}

// Main kernel: streaming S = F F^T / TEMP via split-f16 WMMA, fused with the
// two additive row reductions Z_i and T_i.
// Grid (B/64, 8), block 256 (8 waves). Wave w: row slab wr=w&1 (32 rows),
// col slab wc=w>>1 (32 cols) -> 4 16x16 accumulators, 32x32 outputs/wave.
// 16 b128 loads feed 12 WMMAs per K=32 step (1.33 loads/WMMA).
__global__ void __launch_bounds__(256)
k_main(const _Float16* __restrict__ Fhi, const _Float16* __restrict__ Flo,
       const float* __restrict__ Ww, const float* __restrict__ mhat,
       const int* __restrict__ labels,
       float* __restrict__ gZp, float* __restrict__ gTp) {
    const int lane = threadIdx.x & 31;
    const int wv   = threadIdx.x >> 5;
    const int wr   = wv & 1;      // row slab within block (0..1)
    const int wc   = wv >> 1;     // col slab within 128-wide j-tile (0..3)
    const int n    = lane & 15;
    const int h    = lane >> 4;
    const int i0   = blockIdx.x * 64 + wr * 32;

    float zacc[2][8] = {};
    float tacc[2][8] = {};

    // A-fragments: 16-bit A 16x32 layout -> lane holds row (lane&15),
    // K runs at h*8 and 16+h*8. Two row tiles: i0 and i0+16.
    const size_t rowA0 = (size_t)(i0 + n) * D_;
    const size_t rowA1 = (size_t)(i0 + 16 + n) * D_;
    const _Float16* pA0h = Fhi + rowA0 + h * 8;
    const _Float16* pA0l = Flo + rowA0 + h * 8;
    const _Float16* pA1h = Fhi + rowA1 + h * 8;
    const _Float16* pA1l = Flo + rowA1 + h * 8;

    const float invT = 1.0f / TEMP;

    for (int jt = 0; jt < 4; ++jt) {
        const int j0 = blockIdx.y * 512 + jt * 128 + wc * 32;
        // B-fragments: 16-bit B 32x16 layout -> lane holds column (lane&15),
        // K runs at h*8 and 16+h*8 (matching A). For X·Xᵀ this is a row of F.
        const size_t rowB0 = (size_t)(j0 + n) * D_;
        const size_t rowB1 = (size_t)(j0 + 16 + n) * D_;
        const _Float16* pB0h = Fhi + rowB0 + h * 8;
        const _Float16* pB0l = Flo + rowB0 + h * 8;
        const _Float16* pB1h = Fhi + rowB1 + h * 8;
        const _Float16* pB1l = Flo + rowB1 + h * 8;

        v8f acc00 = {0,0,0,0,0,0,0,0};   // rows i0..i0+15   x cols j0..j0+15
        v8f acc01 = {0,0,0,0,0,0,0,0};   // rows i0..i0+15   x cols j0+16..j0+31
        v8f acc10 = {0,0,0,0,0,0,0,0};   // rows i0+16..i0+31 x cols j0..j0+15
        v8f acc11 = {0,0,0,0,0,0,0,0};   // rows i0+16..i0+31 x cols j0+16..j0+31

#pragma unroll 1
        for (int k = 0; k < D_; k += 32) {
            v16h A0h = packAB(*(const v8h*)(pA0h + k), *(const v8h*)(pA0h + k + 16));
            v16h A0l = packAB(*(const v8h*)(pA0l + k), *(const v8h*)(pA0l + k + 16));
            v16h A1h = packAB(*(const v8h*)(pA1h + k), *(const v8h*)(pA1h + k + 16));
            v16h A1l = packAB(*(const v8h*)(pA1l + k), *(const v8h*)(pA1l + k + 16));
            v16h B0h = packAB(*(const v8h*)(pB0h + k), *(const v8h*)(pB0h + k + 16));
            v16h B0l = packAB(*(const v8h*)(pB0l + k), *(const v8h*)(pB0l + k + 16));
            v16h B1h = packAB(*(const v8h*)(pB1h + k), *(const v8h*)(pB1h + k + 16));
            v16h B1l = packAB(*(const v8h*)(pB1l + k), *(const v8h*)(pB1l + k + 16));
            // S ≈ hi·hi + hi·lo + lo·hi ; 4 independent acc chains interleaved
            // so each acc's RAW chain is 4 WMMAs apart (XDL latency hiding).
            acc00 = __builtin_amdgcn_wmma_f32_16x16x32_f16(false, A0h, false, B0h, (short)0, acc00, false, false);
            acc01 = __builtin_amdgcn_wmma_f32_16x16x32_f16(false, A0h, false, B1h, (short)0, acc01, false, false);
            acc10 = __builtin_amdgcn_wmma_f32_16x16x32_f16(false, A1h, false, B0h, (short)0, acc10, false, false);
            acc11 = __builtin_amdgcn_wmma_f32_16x16x32_f16(false, A1h, false, B1h, (short)0, acc11, false, false);
            acc00 = __builtin_amdgcn_wmma_f32_16x16x32_f16(false, A0h, false, B0l, (short)0, acc00, false, false);
            acc01 = __builtin_amdgcn_wmma_f32_16x16x32_f16(false, A0h, false, B1l, (short)0, acc01, false, false);
            acc10 = __builtin_amdgcn_wmma_f32_16x16x32_f16(false, A1h, false, B0l, (short)0, acc10, false, false);
            acc11 = __builtin_amdgcn_wmma_f32_16x16x32_f16(false, A1h, false, B1l, (short)0, acc11, false, false);
            acc00 = __builtin_amdgcn_wmma_f32_16x16x32_f16(false, A0l, false, B0h, (short)0, acc00, false, false);
            acc01 = __builtin_amdgcn_wmma_f32_16x16x32_f16(false, A0l, false, B1h, (short)0, acc01, false, false);
            acc10 = __builtin_amdgcn_wmma_f32_16x16x32_f16(false, A1l, false, B0h, (short)0, acc10, false, false);
            acc11 = __builtin_amdgcn_wmma_f32_16x16x32_f16(false, A1l, false, B1h, (short)0, acc11, false, false);
        }

        // Fused epilogue (runs once per 32 K-steps; metadata gathered here to
        // keep registers free for the GEMM). C-layout: element (v, lane) is
        // row v+8h, col lane&15.
        const int jA = j0 + n, jB = j0 + 16 + n;
        const int ljA = labels[jA], ljB = labels[jB];
#pragma unroll
        for (int rt = 0; rt < 2; ++rt) {
#pragma unroll
            for (int v = 0; v < 8; ++v) {
                const int   i  = i0 + rt * 16 + v + 8 * h;
                const int   li = labels[i];
                const float di = mhat[i];
                const float sA = (rt ? acc10[v] : acc00[v]) * invT;
                const float sB = (rt ? acc11[v] : acc01[v]) * invT;
                if (jA != i) {
                    zacc[rt][v] += __expf(sA - di) * Ww[(size_t)i * C_ + ljA];
                    if (ljA == li) tacc[rt][v] += sA;
                }
                if (jB != i) {
                    zacc[rt][v] += __expf(sB - di) * Ww[(size_t)i * C_ + ljB];
                    if (ljB == li) tacc[rt][v] += sB;
                }
            }
        }
    }

    // Reduce across the 16 lanes of each half, then write this block's
    // per-row partial to a dedicated slot (deterministic, no atomics).
    const int slot = blockIdx.y * 4 + wc;   // 0..31
#pragma unroll
    for (int rt = 0; rt < 2; ++rt) {
#pragma unroll
        for (int v = 0; v < 8; ++v) {
            float z = zacc[rt][v], t = tacc[rt][v];
#pragma unroll
            for (int o = 8; o >= 1; o >>= 1) {
                z += __shfl_xor(z, o, 16);
                t += __shfl_xor(t, o, 16);
            }
            if (n == 0) {
                const int i = i0 + rt * 16 + v + 8 * h;
                gZp[(size_t)i * 32 + slot] = z;
                gTp[(size_t)i * 32 + slot] = t;
            }
        }
    }
}

// Finisher: merge 32 slots per row, apply the closed-form per-row loss, reduce.
__global__ void k_final(const float* __restrict__ Ww, const float* __restrict__ mhat,
                        const float* __restrict__ gZp, const float* __restrict__ gTp,
                        const int* __restrict__ labels, const int* __restrict__ cnt,
                        float* __restrict__ out) {
    __shared__ float red[256];
    float s = 0.f;
    for (int i = threadIdx.x; i < B_; i += 256) {
        float Zi = 0.f, Ti = 0.f;
#pragma unroll
        for (int sl = 0; sl < 32; ++sl) {
            Zi += gZp[(size_t)i * 32 + sl];
            Ti += gTp[(size_t)i * 32 + sl];
        }
        int   li  = labels[i];
        float Ni  = (float)(cnt[li] - 1);
        float wii = Ww[(size_t)i * C_ + li];
        // mean_log_prob_pos[i] = w_ii * (T_i/N_i - m̂_i - log Z_i)
        s += wii * (Ti / Ni - mhat[i] - __logf(Zi));
    }
    red[threadIdx.x] = s;
    for (int o = 128; o >= 1; o >>= 1) {
        __syncthreads();
        if (threadIdx.x < o) red[threadIdx.x] += red[threadIdx.x + o];
    }
    if (threadIdx.x == 0) out[0] = -red[0] / (float)B_;
}

// ---------------------------------------------------------------------------
extern "C" void kernel_launch(void* const* d_in, const int* in_sizes, int n_in,
                              void* d_out, int out_size, void* d_ws, size_t ws_size,
                              hipStream_t stream) {
    const float* F      = (const float*)d_in[0];   // [B, D] f32
    const int*   labels = (const int*)  d_in[1];   // [B]    int
    const float* Wt     = (const float*)d_in[2];   // [B, C] f32
    float*       out    = (float*)d_out;

    // Workspace layout (~17.6 MB total, all 16B-aligned):
    char* ws = (char*)d_ws;
    _Float16* Fhi  = (_Float16*)(ws);                  //  8 MB f16 hi
    _Float16* Flo  = (_Float16*)(ws + (8ull  << 20));  //  8 MB f16 lo
    float*    Ww   = (float*)   (ws + (16ull << 20));  // 512 KB softmax(w)
    float*    mhat = Ww  + (size_t)B_ * C_;            // 16 KB diag stabilizer
    float*    gZp  = mhat + B_;                        // 512 KB Z partials [B][32]
    float*    gTp  = gZp + (size_t)B_ * 32;            // 512 KB T partials [B][32]
    int*      cnt  = (int*)(gTp + (size_t)B_ * 32);    // 128 B label histogram

    k_zero   <<<dim3(1),                     dim3(C_),  0, stream>>>(cnt);
    k_convert<<<dim3((B_ * D_ + 255) / 256), dim3(256), 0, stream>>>(F, Fhi, Flo);
    k_softmax<<<dim3((B_ + 255) / 256),      dim3(256), 0, stream>>>(Wt, Ww);
    k_diag   <<<dim3(B_ / 8),                dim3(256), 0, stream>>>(F, labels, mhat, cnt);
    k_main   <<<dim3(B_ / 64, 8),            dim3(256), 0, stream>>>(Fhi, Flo, Ww, mhat, labels, gZp, gTp);
    k_final  <<<dim3(1),                     dim3(256), 0, stream>>>(Ww, mhat, gZp, gTp, labels, cnt, out);
}
